// TopKPoolingBlock_33500744909411
// MI455X (gfx1250) — matrix operand order repaired and checked by reading, not compiled, not run on hardware
//
#include <hip/hip_runtime.h>
#include <hip/hip_bf16.h>
#include <math.h>

// Sizes fixed by the reference
#define DD   128       // feature dim
#define NG   256       // graphs
#define EE   524288    // edges (constant across layers, masked)
#define LL   3         // layers
#define OUTS (LL*3*DD) // 1152 output cols per graph

typedef unsigned short u16t;
typedef unsigned int   u32t;
typedef __attribute__((ext_vector_type(16))) __bf16 v16bf;
typedef __attribute__((ext_vector_type(8)))  float  v8f;

union Frag { uint4 u[2]; v16bf v; };

__device__ __forceinline__ u16t f2bf(float f) {
  u32t u = __float_as_uint(f);
  u32t r = u + 0x7FFFu + ((u >> 16) & 1u);   // RNE
  return (u16t)(r >> 16);
}

// ---------- weight packing: B-matrix WMMA layout, fp32 -> bf16 ----------
// wpack[((mat*32 + nt*4 + kc)*32 + lane)*16 + j]
// lane<16: col = nt*16+lane, K = kc*32 + j        (j=0..15)
// lane>=16: col = nt*16+lane-16, K = kc*32 + 16+j
__global__ void k_pack(const float* Wq, const float* Wk, const float* Wv,
                       const float* Ws, const float* Wt, u16t* wpack) {
  int t = blockIdx.x * blockDim.x + threadIdx.x;
  if (t >= 15 * 32 * 32 * 16) return;
  int j = t & 15, lane = (t >> 4) & 31, c = (t >> 9) & 31, mat = t >> 14;
  int kc = c & 3, nt = c >> 2;
  int l = mat / 5, wi = mat % 5;
  const float* W;
  switch (wi) { case 0: W = Wq; break; case 1: W = Wk; break;
                case 2: W = Wv; break; case 3: W = Ws; break; default: W = Wt; }
  W += (size_t)l * DD * DD;
  int col = nt * 16 + (lane & 15);
  int kk  = kc * 32 + ((lane < 16) ? j : 16 + j);
  wpack[t] = f2bf(W[(size_t)kk * DD + col]);
}

__global__ void k_x2bf(const float* x, u16t* xbf, int n) {
  int i = blockIdx.x * blockDim.x + threadIdx.x;
  if (i < n) xbf[i] = f2bf(x[i]);
}

__global__ void k_edges_init(const int* ei, int* srcb, int* dstb, int* maskb) {
  int e = blockIdx.x * blockDim.x + threadIdx.x;
  if (e < EE) { srcb[e] = ei[e]; dstb[e] = ei[EE + e]; maskb[e] = 1; }
}

// ---------- WMMA GEMM: out = X[Nc,128] @ W + b, 4 waves = {Q,K,V,skip} ----------
__global__ void k_gemm_qkvs(const u16t* __restrict__ xbf, const u16t* __restrict__ wpl,
                            const float* bq, const float* bk, const float* bv, const float* bs,
                            float* q, float* k, float* v, float* s) {
  int wv_id = threadIdx.x >> 5;
  int lane  = threadIdx.x & 31;
  int half  = lane >> 4;
  int m0    = blockIdx.x * 16;
  const float* bias; float* out;
  switch (wv_id) { case 0: bias = bq; out = q; break; case 1: bias = bk; out = k; break;
                   case 2: bias = bv; out = v; break; default: bias = bs; out = s; }
  const u16t* wp = wpl + wv_id * (32 * 32 * 16);

  Frag A[4];
  int row = m0 + (lane & 15);
  const u16t* xr = xbf + (size_t)row * DD + half * 16;
#pragma unroll
  for (int kc = 0; kc < 4; kc++) {
    const u16t* p = xr + kc * 32;
    A[kc].u[0] = *(const uint4*)p;          // K-lo chunk
    A[kc].u[1] = *(const uint4*)(p + 8);   // K-hi chunk
  }
  int col = lane & 15;
#pragma unroll
  for (int nt = 0; nt < 8; nt++) {
    v8f acc = {0.f, 0.f, 0.f, 0.f, 0.f, 0.f, 0.f, 0.f};
#pragma unroll
    for (int kc = 0; kc < 4; kc++) {
      Frag B;
      const u16t* bp = wp + (((nt * 4 + kc) * 32 + lane) << 4);
      B.u[0] = *(const uint4*)bp;
      B.u[1] = *(const uint4*)(bp + 8);
      acc = __builtin_amdgcn_wmma_f32_16x16x32_bf16(false, A[kc].v, false, B.v,
                                                    (short)0, acc, false, false);
    }
    int n0 = nt * 16;
    float b = bias[n0 + col];
    int rb = m0 + half * 8;
#pragma unroll
    for (int r = 0; r < 8; r++)
      out[(size_t)(rb + r) * DD + n0 + col] = acc[r] + b;
  }
}

// Single-matrix GEMM + ReLU, 4 waves = 4 row strips
__global__ void k_gemm_relu(const u16t* __restrict__ xbf, const u16t* __restrict__ wp,
                            const float* bias, float* out) {
  int wv_id = threadIdx.x >> 5;
  int lane  = threadIdx.x & 31;
  int half  = lane >> 4;
  int m0    = blockIdx.x * 64 + wv_id * 16;

  Frag A[4];
  int row = m0 + (lane & 15);
  const u16t* xr = xbf + (size_t)row * DD + half * 16;
#pragma unroll
  for (int kc = 0; kc < 4; kc++) {
    const u16t* p = xr + kc * 32;
    A[kc].u[0] = *(const uint4*)p;
    A[kc].u[1] = *(const uint4*)(p + 8);
  }
  int col = lane & 15;
#pragma unroll
  for (int nt = 0; nt < 8; nt++) {
    v8f acc = {0.f, 0.f, 0.f, 0.f, 0.f, 0.f, 0.f, 0.f};
#pragma unroll
    for (int kc = 0; kc < 4; kc++) {
      Frag B;
      const u16t* bp = wp + (((nt * 4 + kc) * 32 + lane) << 4);
      B.u[0] = *(const uint4*)bp;
      B.u[1] = *(const uint4*)(bp + 8);
      acc = __builtin_amdgcn_wmma_f32_16x16x32_bf16(false, A[kc].v, false, B.v,
                                                    (short)0, acc, false, false);
    }
    int n0 = nt * 16;
    float b = bias[n0 + col];
    int rb = m0 + half * 8;
#pragma unroll
    for (int r = 0; r < 8; r++)
      out[(size_t)(rb + r) * DD + n0 + col] = fmaxf(acc[r] + b, 0.f);
  }
}

// ---------- edge attention: logits, then CSR build + fused softmax-aggregate ----------
__global__ void k_logits(const float* __restrict__ q, const float* __restrict__ k,
                         const int* srcb, const int* dstb, const int* maskb,
                         float* logits) {
  int wid  = (blockIdx.x * blockDim.x + threadIdx.x) >> 5;
  int lane = threadIdx.x & 31;
  if (wid >= EE) return;
  if (!maskb[wid]) return;
  int s = srcb[wid], d = dstb[wid];
  float4 q4 = *(const float4*)(q + (size_t)d * DD + lane * 4);
  float4 k4 = *(const float4*)(k + (size_t)s * DD + lane * 4);
  float p = q4.x * k4.x + q4.y * k4.y + q4.z * k4.z + q4.w * k4.w;
  for (int off = 16; off; off >>= 1) p += __shfl_xor(p, off, 32);
  if (lane == 0) logits[wid] = p * 0.08838834764831845f;   // 1/sqrt(128)
}

__global__ void k_init_csr(int* cnt, int* fillc, int nc) {
  int i = blockIdx.x * blockDim.x + threadIdx.x;
  if (i < nc) { cnt[i] = 0; fillc[i] = 0; }
}

__global__ void k_indeg(const int* dstb, const int* maskb, int* cnt) {
  int e = blockIdx.x * blockDim.x + threadIdx.x;
  if (e >= EE) return;
  if (maskb[e]) atomicAdd(cnt + dstb[e], 1);
}

// exclusive scan of cnt -> offs (two-level, 256-wide chunks)
__global__ void k_scan_block(const int* cnt, int* offs, int* bsum) {
  int t = threadIdx.x;
  int base = blockIdx.x * 256;
  __shared__ int s[256];
  int mine = cnt[base + t];
  s[t] = mine; __syncthreads();
  for (int o = 1; o < 256; o <<= 1) {
    int add = (t >= o) ? s[t - o] : 0;
    __syncthreads();
    s[t] += add;
    __syncthreads();
  }
  offs[base + t] = s[t] - mine;
  if (t == 255) bsum[blockIdx.x] = s[255];
}

__global__ void k_scan_top(const int* bsum, int* boff, int nb) {
  int t = threadIdx.x;
  __shared__ int s[256];
  int mine = (t < nb) ? bsum[t] : 0;
  s[t] = mine; __syncthreads();
  for (int o = 1; o < 256; o <<= 1) {
    int add = (t >= o) ? s[t - o] : 0;
    __syncthreads();
    s[t] += add;
    __syncthreads();
  }
  if (t < nb) boff[t] = s[t] - mine;
}

__global__ void k_scan_add(int* offs, const int* boff, int nc) {
  int i = blockIdx.x * blockDim.x + threadIdx.x;
  if (i < nc) offs[i] += boff[i >> 8];
}

__global__ void k_fill(const int* dstb, const int* maskb, const int* offs,
                       int* fillc, int* elist) {
  int e = blockIdx.x * blockDim.x + threadIdx.x;
  if (e >= EE) return;
  if (!maskb[e]) return;
  int d = dstb[e];
  int p = atomicAdd(fillc + d, 1);
  elist[offs[d] + p] = e;
}

// one wave per dst node: fused segment max + exp-sum + weighted gather of v
__global__ void k_aggregate(const float* __restrict__ logits, const float* __restrict__ v,
                            const int* srcb, const int* elist, const int* offs,
                            const int* cnt, float* agg, int nc) {
  int d    = (blockIdx.x * blockDim.x + threadIdx.x) >> 5;
  int lane = threadIdx.x & 31;
  if (d >= nc) return;
  int beg = offs[d], deg = cnt[d];
  float mx = -3.4e38f;
  for (int i = lane; i < deg; i += 32) mx = fmaxf(mx, logits[elist[beg + i]]);
  for (int off = 16; off; off >>= 1) mx = fmaxf(mx, __shfl_xor(mx, off, 32));
  float4 acc = {0.f, 0.f, 0.f, 0.f};
  float s = 0.f;
  int c0 = lane * 4;
  for (int i = 0; i < deg; i++) {
    int e = elist[beg + i];
    float w = expf(logits[e] - mx);
    s += w;
    float4 v4 = *(const float4*)(v + (size_t)srcb[e] * DD + c0);
    acc.x += w * v4.x; acc.y += w * v4.y; acc.z += w * v4.z; acc.w += w * v4.w;
  }
  float inv = (deg > 0) ? 1.f / fmaxf(s, 1e-16f) : 0.f;
  acc.x *= inv; acc.y *= inv; acc.z *= inv; acc.w *= inv;
  *(float4*)(agg + (size_t)d * DD + c0) = acc;
}

__global__ void k_addskip(const float* agg, const float* skip, u16t* xbf, int tot) {
  int i = blockIdx.x * blockDim.x + threadIdx.x;
  if (i < tot) xbf[i] = f2bf(agg[i] + skip[i]);
}

// ---------- batchnorm + score ----------
__global__ void k_bnstats(const float* y, float* mu, float* rstd, int nc) {
  int c = blockIdx.x, t = threadIdx.x;
  float s = 0.f, s2 = 0.f;
  for (int r = t; r < nc; r += blockDim.x) {
    float v = y[(size_t)r * DD + c]; s += v; s2 += v * v;
  }
  __shared__ float ss[256], ssq[256];
  ss[t] = s; ssq[t] = s2; __syncthreads();
  for (int o = 128; o; o >>= 1) {
    if (t < o) { ss[t] += ss[t + o]; ssq[t] += ssq[t + o]; }
    __syncthreads();
  }
  if (t == 0) {
    float mn = ss[0] / nc;
    float var = ssq[0] / nc - mn * mn;
    mu[c] = mn;
    rstd[c] = rsqrtf(fmaxf(var, 0.f) + 1e-5f);
  }
}

__global__ void k_poolnorm(const float* pw, float* normw) {
  int t = threadIdx.x; // 128
  __shared__ float ss[128];
  float w = pw[t];
  ss[t] = w * w; __syncthreads();
  for (int o = 64; o; o >>= 1) { if (t < o) ss[t] += ss[t + o]; __syncthreads(); }
  normw[t] = w * rsqrtf(ss[0]);
}

__global__ void k_bnscore(float* y, const float* mu, const float* rstd,
                          const float* gamma, const float* beta,
                          const float* normw, float* score, int nc) {
  int wid  = (blockIdx.x * blockDim.x + threadIdx.x) >> 5;
  int lane = threadIdx.x & 31;
  if (wid >= nc) return;
  int c0 = lane * 4;
  float4 v  = *(float4*)(y + (size_t)wid * DD + c0);
  float4 m4 = *(const float4*)(mu + c0);
  float4 r4 = *(const float4*)(rstd + c0);
  float4 g4 = *(const float4*)(gamma + c0);
  float4 b4 = *(const float4*)(beta + c0);
  float4 w4 = *(const float4*)(normw + c0);
  v.x = (v.x - m4.x) * r4.x * g4.x + b4.x;
  v.y = (v.y - m4.y) * r4.y * g4.y + b4.y;
  v.z = (v.z - m4.z) * r4.z * g4.z + b4.z;
  v.w = (v.w - m4.w) * r4.w * g4.w + b4.w;
  *(float4*)(y + (size_t)wid * DD + c0) = v;
  float p = v.x * w4.x + v.y * w4.y + v.z * w4.z + v.w * w4.w;
  for (int off = 16; off; off >>= 1) p += __shfl_xor(p, off, 32);
  if (lane == 0) score[wid] = tanhf(p);
}

// ---------- top-k per graph (O(n^2) rank, tie-break by index) ----------
__global__ void k_topk(const float* score, int* newid, int* perm_old, int n, int kk) {
  int g = blockIdx.x, t = threadIdx.x;
  __shared__ float s[256];
  __shared__ int f[256];
  float mys = score[g * n + t];
  s[t] = mys; __syncthreads();
  int rank = 0;
  for (int j = 0; j < n; j++) {
    float sj = s[j];
    rank += (sj > mys) || (sj == mys && j < t);
  }
  int sel = rank < kk;
  f[t] = sel; __syncthreads();
  int pos = 0;
  for (int j = 0; j < t; j++) pos += f[j];
  int old = g * n + t;
  if (sel) { int ni = g * kk + pos; newid[old] = ni; perm_old[ni] = old; }
  else newid[old] = -1;
}

__global__ void k_gather(const float* y, const float* score, const int* perm_old,
                         float* xnew, u16t* xbf, int nnew) {
  int wid  = (blockIdx.x * blockDim.x + threadIdx.x) >> 5;
  int lane = threadIdx.x & 31;
  if (wid >= nnew) return;
  int old = perm_old[wid];
  float sc = score[old];
  int c0 = lane * 4;
  float4 v = *(const float4*)(y + (size_t)old * DD + c0);
  v.x *= sc; v.y *= sc; v.z *= sc; v.w *= sc;
  *(float4*)(xnew + (size_t)wid * DD + c0) = v;
  ushort4 bf;
  bf.x = f2bf(v.x); bf.y = f2bf(v.y); bf.z = f2bf(v.z); bf.w = f2bf(v.w);
  *(ushort4*)(xbf + (size_t)wid * DD + c0) = bf;
}

__global__ void k_reindex(const int* newid, int* srcb, int* dstb, int* maskb) {
  int e = blockIdx.x * blockDim.x + threadIdx.x;
  if (e >= EE) return;
  int mk = maskb[e];
  int s2 = newid[srcb[e]], d2 = newid[dstb[e]];
  mk = mk && (s2 >= 0) && (d2 >= 0);
  maskb[e] = mk;
  srcb[e] = mk ? s2 : 0;
  dstb[e] = mk ? d2 : 0;
}

__global__ void k_readout(const float* xnew, float* out, int kk, int loff) {
  int g = blockIdx.x, c = threadIdx.x; // 128 threads
  float mx = -3.4e38f, sm = 0.f;
  const float* p = xnew + (size_t)g * kk * DD + c;
  for (int j = 0; j < kk; j++) {
    float v = p[(size_t)j * DD];
    mx = fmaxf(mx, v); sm += v;
  }
  float* o = out + (size_t)g * OUTS + loff;
  o[c] = mx;
  o[DD + c] = sm / (float)kk;
  o[2 * DD + c] = sm;
}

// ------------------------------------------------------------------
extern "C" void kernel_launch(void* const* d_in, const int* in_sizes, int n_in,
                              void* d_out, int out_size, void* d_ws, size_t ws_size,
                              hipStream_t stream) {
  const float* x    = (const float*)d_in[0];
  const int*   ei   = (const int*)d_in[1];
  const float* Wq   = (const float*)d_in[4];
  const float* bq   = (const float*)d_in[5];
  const float* Wk   = (const float*)d_in[6];
  const float* bk   = (const float*)d_in[7];
  const float* Wv   = (const float*)d_in[8];
  const float* bv   = (const float*)d_in[9];
  const float* Wsk  = (const float*)d_in[10];
  const float* bsk  = (const float*)d_in[11];
  const float* Wt   = (const float*)d_in[12];
  const float* bt   = (const float*)d_in[13];
  const float* gam  = (const float*)d_in[14];
  const float* bet  = (const float*)d_in[15];
  const float* pw   = (const float*)d_in[16];

  const int NN0 = 65536;
  char* base = (char*)d_ws;
  size_t off = 0;
  auto carve = [&](size_t bytes) -> char* {
    char* p = base + off;
    off = (off + bytes + 255) & ~(size_t)255;
    return p;
  };
  u16t*  wpack = (u16t*)carve((size_t)15 * 16384 * 2);
  u16t*  xbf   = (u16t*)carve((size_t)NN0 * DD * 2);
  float* q     = (float*)carve((size_t)NN0 * DD * 4);
  float* k     = (float*)carve((size_t)NN0 * DD * 4);
  float* v     = (float*)carve((size_t)NN0 * DD * 4);
  float* skip  = (float*)carve((size_t)NN0 * DD * 4);
  float* agg   = (float*)carve((size_t)NN0 * DD * 4);
  float* y     = (float*)carve((size_t)NN0 * DD * 4);
  float* xnew  = (float*)carve((size_t)(NN0 / 2) * DD * 4);
  float* logits= (float*)carve((size_t)EE * 4);
  float* score = (float*)carve((size_t)NN0 * 4);
  float* mu    = (float*)carve(512);
  float* rstd  = (float*)carve(512);
  float* normw = (float*)carve(512);
  int*   newid = (int*)carve((size_t)NN0 * 4);
  int*   permo = (int*)carve((size_t)(NN0 / 2) * 4);
  int*   srcb  = (int*)carve((size_t)EE * 4);
  int*   dstb  = (int*)carve((size_t)EE * 4);
  int*   maskb = (int*)carve((size_t)EE * 4);
  int*   cnt   = (int*)carve((size_t)NN0 * 4);
  int*   fillc = (int*)carve((size_t)NN0 * 4);
  int*   offsb = (int*)carve((size_t)NN0 * 4);
  int*   bsum  = (int*)carve(256 * 4);
  int*   boff  = (int*)carve(256 * 4);
  int*   elist = (int*)carve((size_t)EE * 4);

  k_pack<<<(15 * 32 * 32 * 16 + 255) / 256, 256, 0, stream>>>(Wq, Wk, Wv, Wsk, Wt, wpack);
  k_x2bf<<<(NN0 * DD) / 256, 256, 0, stream>>>(x, xbf, NN0 * DD);
  k_edges_init<<<EE / 256, 256, 0, stream>>>(ei, srcb, dstb, maskb);

  int nc = NN0, n = 256;
  for (int l = 0; l < LL; l++) {
    int kk = n / 2;
    int nnew = NG * kk;
    int nb = nc / 256;
    const u16t* wpl = wpack + (size_t)l * 5 * 16384;

    k_gemm_qkvs<<<nc / 16, 128, 0, stream>>>(xbf, wpl,
        bq + l * DD, bk + l * DD, bv + l * DD, bsk + l * DD, q, k, v, skip);
    k_logits<<<EE / 8, 256, 0, stream>>>(q, k, srcb, dstb, maskb, logits);
    // CSR build
    k_init_csr<<<nc / 256, 256, 0, stream>>>(cnt, fillc, nc);
    k_indeg<<<EE / 256, 256, 0, stream>>>(dstb, maskb, cnt);
    k_scan_block<<<nb, 256, 0, stream>>>(cnt, offsb, bsum);
    k_scan_top<<<1, 256, 0, stream>>>(bsum, boff, nb);
    k_scan_add<<<nc / 256, 256, 0, stream>>>(offsb, boff, nc);
    k_fill<<<EE / 256, 256, 0, stream>>>(dstb, maskb, offsb, fillc, elist);
    // fused softmax-aggregate (no fp atomics)
    k_aggregate<<<nc / 8, 256, 0, stream>>>(logits, v, srcb, elist, offsb, cnt, agg, nc);

    k_addskip<<<(nc * DD) / 256, 256, 0, stream>>>(agg, skip, xbf, nc * DD);
    k_gemm_relu<<<nc / 64, 128, 0, stream>>>(xbf, wpl + 4 * 16384, bt + l * DD, y);
    k_bnstats<<<DD, 256, 0, stream>>>(y, mu, rstd, nc);
    k_poolnorm<<<1, DD, 0, stream>>>(pw + l * DD, normw);
    k_bnscore<<<nc / 8, 256, 0, stream>>>(y, mu, rstd, gam + l * DD, bet + l * DD,
                                          normw, score, nc);
    k_topk<<<NG, n, 0, stream>>>(score, newid, permo, n, kk);
    k_gather<<<nnew / 8, 256, 0, stream>>>(y, score, permo, xnew, xbf, nnew);
    k_reindex<<<EE / 256, 256, 0, stream>>>(newid, srcb, dstb, maskb);
    k_readout<<<NG, DD, 0, stream>>>(xnew, (float*)d_out, kk, l * 3 * DD);

    nc = nnew;
    n = kk;
  }
  (void)in_sizes; (void)n_in; (void)out_size; (void)ws_size;
}
